// DomainGate_68908455297139
// MI455X (gfx1250) — matrix operand order repaired and checked by reading, not compiled, not run on hardware
//
#include <hip/hip_runtime.h>

// ---------------------------------------------------------------------------
// DomainGate for MI455X (gfx1250, wave32).
// T=8192 tokens, E=16 experts, C=512 capacity.
// Output layout (f32): [0] l_aux, [1 .. 1+TEC) combine, [1+TEC .. 1+2*TEC) mask
// TEC = 8192*16*512 = 67,108,864.  Total ~537MB -> store-bandwidth bound
// (~23us @ 23.3 TB/s). Scan work is done with a WMMA triangular-matmul
// prefix sum (one v_wmma_f32_16x16x32_f16 per 16-token tile).
// ---------------------------------------------------------------------------

typedef __attribute__((ext_vector_type(16))) _Float16 v16h;
typedef __attribute__((ext_vector_type(8)))  float    v8f;
typedef __attribute__((ext_vector_type(4)))  float    v4f;

#define T_TOK  8192
#define NEXP   16
#define CAP    512                      // ceil(T/E)
#define NTILE  (T_TOK / 16)             // 512 tiles of 16 tokens
#define TEC    ((long long)T_TOK * NEXP * CAP)

// ---------------------------------------------------------------------------
// Kernel 1: zero the entire output with non-temporal b128 streaming stores.
// 537MB >> 192MB L2, so NT avoids dirtying the cache hierarchy.
// ---------------------------------------------------------------------------
__global__ __launch_bounds__(256) void fill_zero_kernel(float* __restrict__ out,
                                                        long long n) {
  const v4f z = {0.0f, 0.0f, 0.0f, 0.0f};
  const long long n4 = n >> 2;
  long long i = (long long)blockIdx.x * blockDim.x + threadIdx.x;
  const long long stride = (long long)gridDim.x * blockDim.x;
  v4f* __restrict__ o4 = (v4f*)out;
  for (long long v = i; v < n4; v += stride)
    __builtin_nontemporal_store(z, o4 + v);
  if (i == 0)  // scalar tail (out_size = 1 + 2*TEC is odd)
    for (long long k = n4 << 2; k < n; ++k)
      __builtin_nontemporal_store(0.0f, out + k);
}

// ---------------------------------------------------------------------------
// Kernel 2: single-workgroup gate scan (8 waves).
//   Pass 1: per 16-token tile, per-expert counts via WMMA:
//           P = L16 x M, L16 = lower-triangular ones (16x32 f16, K>=16 zero),
//           M = 16(token-K) x 16(expert-N) keep flags. Column totals = row 15.
//   Pass 2: serial exclusive scan of tile counts per expert (512 steps, tiny).
//   Pass 3: per-token slot via wave32 match-any ballots + tile base;
//           emit flat output index or -1 into workspace.
// ---------------------------------------------------------------------------
__global__ __launch_bounds__(256) void gate_scan_kernel(
    const int* __restrict__ domain_ids,
    const unsigned char* __restrict__ pad_mask,   // jax bool: 1 byte/elem
    int* __restrict__ rec) {
  __shared__ unsigned short sTileCnt[NTILE * NEXP];   // 16KB (counts <= 16)
  __shared__ int            sTileBase[NTILE * NEXP];  // 32KB

  const int tid  = threadIdx.x;
  const int lane = tid & 31;
  const int wv   = tid >> 5;            // wave 0..7
  const int NW   = blockDim.x >> 5;

  // Lower-triangular A operand (16x32 f16; only K<16 populated).
  // A, contiguous k order used here: lanes 0-15 hold row m=lane, elems 0..15 <-> K=0..15;
  // lanes 16-31 hold row m=lane-16, elems 0..15 <-> K=16..31 (zero padding).
  v16h A;
  {
    const int m     = lane & 15;
    for (int j = 0; j < 16; ++j) {
      A[j]     = (_Float16)(((lane < 16) && (j <= m)) ? 1.0f : 0.0f);
    }
  }

  // ---- Pass 1: tile counts via WMMA ----
  for (int tile = wv; tile < NTILE; tile += NW) {
    const int t0 = tile * 16;
    int e = 0, keep = 0;
    if (lane < 16) {
      e    = domain_ids[t0 + lane];
      keep = pad_mask[t0 + lane] ? 0 : 1;
    }
    const int packed = (keep << 8) | (e & 0xFF);

    // B operand (32x16 f16): lanes 0-15 = expert column n=lane,
    // elem k <-> token k of tile (K=0..15); lanes 16-31 (K=16..31) zero.
    v16h B = {};
    if (lane < 16) {
      for (int k = 0; k < 16; ++k) {
        const int pk = __shfl(packed, k, 32);
        B[k] = (_Float16)(((pk & 0x1FF) == (0x100 | lane)) ? 1.0f : 0.0f);
      }
    }

    v8f Cacc = {};
    Cacc = __builtin_amdgcn_wmma_f32_16x16x32_f16(
        /*neg_a=*/false, A, /*neg_b=*/false, B,
        /*c_mod=*/(short)0, Cacc, /*reuse_a=*/false, /*reuse_b=*/false);

    // D[15][n] (column totals) lives in lanes 16-31, acc component 7.
    if (lane >= 16)
      sTileCnt[tile * NEXP + (lane - 16)] = (unsigned short)(Cacc[7] + 0.5f);
  }
  __syncthreads();

  // ---- Pass 2: exclusive prefix over tiles, one thread per expert ----
  if (tid < NEXP) {
    int run = 0;
    for (int tile = 0; tile < NTILE; ++tile) {
      sTileBase[tile * NEXP + tid] = run;
      run += (int)sTileCnt[tile * NEXP + tid];
    }
  }
  __syncthreads();

  // ---- Pass 3: per-token slot assignment ----
  for (int tile = wv; tile < NTILE; tile += NW) {
    const int t0 = tile * 16;
    int e = 0, keep = 0;
    if (lane < 16) {
      e    = domain_ids[t0 + lane];
      keep = pad_mask[t0 + lane] ? 0 : 1;
    }
    // match-any over the 16 tokens of this tile (lanes 0-15)
    unsigned mymask = 0u;
    for (int v = 0; v < NEXP; ++v) {
      const unsigned bm =
          (unsigned)__ballot((lane < 16) && keep && (e == v));
      if (e == v) mymask = bm;
    }
    if (lane < 16) {
      int r = -1;
      if (keep) {
        const int posInTile = __popc(mymask & ((1u << lane) - 1u));
        const int pos = sTileBase[tile * NEXP + e] + posInTile;
        if (pos < CAP)
          r = (t0 + lane) * (NEXP * CAP) + e * CAP + pos;  // < 2^27, fits i32
      }
      rec[t0 + lane] = r;
    }
  }
}

// ---------------------------------------------------------------------------
// Kernel 3: scatter the ones into combine and dispatch_mask regions.
// ---------------------------------------------------------------------------
__global__ __launch_bounds__(256) void scatter_kernel(
    const int* __restrict__ rec, float* __restrict__ out) {
  const int t = blockIdx.x * blockDim.x + threadIdx.x;
  if (t < T_TOK) {
    const int r = rec[t];
    if (r >= 0) {
      out[1 + (long long)r]       = 1.0f;   // combine1_sec
      out[1 + TEC + (long long)r] = 1.0f;   // dispatch_mask (as 1.0)
    }
  }
}

// ---------------------------------------------------------------------------
extern "C" void kernel_launch(void* const* d_in, const int* in_sizes, int n_in,
                              void* d_out, int out_size, void* d_ws,
                              size_t ws_size, hipStream_t stream) {
  (void)in_sizes; (void)n_in; (void)ws_size;
  // d_in[0] = input [T,D] f32 (unused by the math), d_in[1] = domain_ids i32,
  // d_in[2] = mask (bool, 1 byte per element)
  const int*           domain_ids = (const int*)d_in[1];
  const unsigned char* pad_mask   = (const unsigned char*)d_in[2];
  float*               out        = (float*)d_out;
  int*                 rec        = (int*)d_ws;   // 8192 ints

  // 1) zero-fill ~537MB output (the runtime; store-bandwidth bound)
  const long long n  = (long long)out_size;
  const long long n4 = n >> 2;
  long long blk = (n4 + 256LL * 8 - 1) / (256LL * 8);   // ~8 float4 per thread
  if (blk > 16384) blk = 16384;
  if (blk < 1) blk = 1;
  fill_zero_kernel<<<(int)blk, 256, 0, stream>>>(out, n);

  // 2) WMMA-based gate scan (single WGP; ~40KB traffic, negligible)
  gate_scan_kernel<<<1, 256, 0, stream>>>(domain_ids, pad_mask, rec);

  // 3) scatter <= 8192 ones into both output regions
  scatter_kernel<<<(T_TOK + 255) / 256, 256, 0, stream>>>(rec, out);
}
